// MultiHeadLatentAttention_72387378807199
// MI455X (gfx1250) — matrix operand order repaired and checked by reading, not compiled, not run on hardware
//
#include <hip/hip_runtime.h>
#include <math.h>

// ---------------- CDNA5 / gfx1250 types ----------------
typedef __attribute__((ext_vector_type(16))) __bf16 v16bf;
typedef __attribute__((ext_vector_type(8)))  __bf16 v8bf;
typedef __attribute__((ext_vector_type(8)))  float  v8f;
typedef __attribute__((ext_vector_type(4)))  unsigned int u32x4;
typedef __attribute__((ext_vector_type(8)))  int i32x8;
typedef __attribute__((ext_vector_type(4)))  int i32x4;

#define Hdim 2048
#define NHEAD 16
#define Ddim 128
#define LATD 512
#define Bdim 2
#define Sdim 2048
#define Mrows (Bdim*Sdim)   // 4096

// ---------------- WMMA operand loaders (generic addrspace) ----------------
// A-matrix 16x32 bf16 layout (ISA 7.12.2): lane l -> m=l&15, h=l>>4;
// elems 0..7 = K[h*8 .. h*8+7], elems 8..15 = K[16+h*8 .. 16+h*8+7]
__device__ inline v16bf load_a_16x32(const __bf16* base, int ld) {
    const int l = threadIdx.x & 31;
    const int m = l & 15, h = l >> 4;
    const __bf16* p = base + m * ld + h * 8;
    union { v16bf v; v8bf half2[2]; } u;
    u.half2[0] = *(const v8bf*)(p);
    u.half2[1] = *(const v8bf*)(p + 16);
    return u.v;
}

// B-matrix 32x16 bf16 layout: lane l -> n=l&15, h=l>>4;
// elems 0..7 = K[h*8 .. h*8+7], elems 8..15 = K[16+h*8 .. 16+h*8+7]
__device__ inline v16bf load_b_32x16(const __bf16* base, int ld) {
    const int l = threadIdx.x & 31;
    const int n = l & 15, h = l >> 4;
    v16bf v;
    #pragma unroll
    for (int s = 0; s < 8; ++s) {
        v[s]     = base[(h * 8 + s) * ld + n];
        v[8 + s] = base[(16 + h * 8 + s) * ld + n];
    }
    return v;
}

// A-operand direct from global (row-major [m][k], 16x32 tile)
__device__ inline v16bf load_a_global(const __bf16* rowptr, int koff) {
    const int h = (threadIdx.x & 31) >> 4;
    const __bf16* p = rowptr + koff + h * 8;
    union { v16bf v; v8bf half2[2]; } u;
    u.half2[0] = *(const v8bf*)(p);
    u.half2[1] = *(const v8bf*)(p + 16);
    return u.v;
}

__device__ inline v8f wmma_bf16(v16bf a, v16bf b, v8f c) {
    return __builtin_amdgcn_wmma_f32_16x16x32_bf16(
        /*neg_a=*/false, a, /*neg_b=*/false, b,
        /*c_mod=*/(short)0, c, /*reuse_a=*/false, /*reuse_b=*/false);
}

// ---------------- Tensor Data Mover: 2D bf16 tile -> LDS --------------------
// Builds a D# per CDNA5 ISA and issues TENSOR_LOAD_TO_LDS.
// pad codes: interval 2^(code+1) DWORDs, amount (code+1) DWORDs; matches the
// +8 bf16 row padding of the LDS tiles. Tracked by TENSORcnt.
// Toolchain uses the 6-arg builtin: (g0, g1, g2, g3, gx(8), cpol).
__device__ inline void tdm_load_2d_bf16(unsigned lds_byte_addr, const void* gptr,
                                        int tile_w, int tile_h, int row_stride,
                                        int pad_interval_code, int pad_amount_code) {
    unsigned long long ga = (unsigned long long)(size_t)gptr;
    u32x4 g0;
    g0[0] = 1u;                                   // count=1, user descriptor
    g0[1] = lds_byte_addr;                        // lds_addr
    g0[2] = (unsigned)(ga & 0xffffffffu);         // global_addr[31:0]
    g0[3] = (unsigned)((ga >> 32) & 0x1ffffffu)   // global_addr[56:32]
          | (2u << 30);                           // type = 2 ("image")
    i32x8 g1;
    g1[0] = (int)((1u << 16)                      // data_size = 2 bytes
          | (1u << 20)                            // pad_enable
          | ((unsigned)pad_interval_code << 22)
          | ((unsigned)pad_amount_code  << 25));
    g1[1] = (int)(((unsigned)tile_w & 0xffffu) << 16);   // tensor_dim0[15:0]
    g1[2] = (int)(((unsigned)tile_h & 0xffffu) << 16);   // tensor_dim1[15:0]
    g1[3] = (int)(((unsigned)tile_w & 0xffffu) << 16);   // tile_dim0
    g1[4] = (int)((unsigned)tile_h & 0xffffu);           // tile_dim1 (tile_dim2=0)
    g1[5] = row_stride;                                  // tensor_dim0_stride[31:0]
    g1[6] = 0; g1[7] = 0;                                // dim1 stride unused (2D)
    i32x4 gz  = {0, 0, 0, 0};
    i32x8 gz8 = {0, 0, 0, 0, 0, 0, 0, 0};
    __builtin_amdgcn_tensor_load_to_lds(g0, g1, gz, gz, gz8, 0);
}

// ---------------- Generic tiled GEMM: C[M,N] = A[M,K] @ Bw[K,N] + bias -----
// BM=128 BN=128 BK=32, 256 threads (8 waves), wave tile 64x32.
// bf16-A tiles come in via TDM (overlapped with f32->bf16 B staging).
template<bool A_F32, bool OUT_F32>
__global__ __launch_bounds__(256)
void gemm_wmma_kernel(const void* __restrict__ Ap,
                      const float* __restrict__ Bw,
                      const float* __restrict__ bias,
                      void* __restrict__ Cp,
                      int M, int N, int K) {
    __shared__ __bf16 sA[128][40];    // BK=32 + 8 pad (TDM pad: 16 dw interval, 4 dw)
    __shared__ __bf16 sB[32][136];    // BN=128 + 8 pad

    const int tid = threadIdx.x;
    const int n0 = blockIdx.x * 128;
    const int m0 = blockIdx.y * 128;

    const int wv   = tid >> 5;
    const int wrow = wv >> 2;
    const int wcol = wv & 3;

    v8f acc[4][2];
    const v8f vzero = {0,0,0,0,0,0,0,0};
    #pragma unroll
    for (int mi = 0; mi < 4; ++mi)
        #pragma unroll
        for (int ni = 0; ni < 2; ++ni) acc[mi][ni] = vzero;

    const float*  Af = (const float*) Ap;
    const __bf16* Ab = (const __bf16*)Ap;

    for (int k0 = 0; k0 < K; k0 += 32) {
        __syncthreads();
        // ---- stage A tile ----
        if (A_F32) {
            const int row = tid >> 1, seg = tid & 1;
            const float* ap = Af + (size_t)(m0 + row) * K + k0 + seg * 16;
            float t[16];
            #pragma unroll
            for (int j = 0; j < 4; ++j) {
                float4 q = ((const float4*)ap)[j];
                t[4*j]=q.x; t[4*j+1]=q.y; t[4*j+2]=q.z; t[4*j+3]=q.w;
            }
            v8bf lo, hi;
            #pragma unroll
            for (int j = 0; j < 8; ++j) { lo[j]=(__bf16)t[j]; hi[j]=(__bf16)t[j+8]; }
            __bf16* dst = &sA[row][seg * 16];
            *(v8bf*)(dst)     = lo;
            *(v8bf*)(dst + 8) = hi;
        } else {
            if (wv == 0)   // async DMA, overlapped with B staging below
                tdm_load_2d_bf16((unsigned)(size_t)&sA[0][0],
                                 Ab + (size_t)m0 * K + k0,
                                 /*w=*/32, /*h=*/128, /*stride=*/K,
                                 /*pad_int=*/3, /*pad_amt=*/3);
        }
        // ---- stage B tile: 32x128 f32 -> bf16 ----
        {
            const int row = tid >> 3, seg = tid & 7;
            const float* bp = Bw + (size_t)(k0 + row) * N + n0 + seg * 16;
            float t[16];
            #pragma unroll
            for (int j = 0; j < 4; ++j) {
                float4 q = ((const float4*)bp)[j];
                t[4*j]=q.x; t[4*j+1]=q.y; t[4*j+2]=q.z; t[4*j+3]=q.w;
            }
            v8bf lo, hi;
            #pragma unroll
            for (int j = 0; j < 8; ++j) { lo[j]=(__bf16)t[j]; hi[j]=(__bf16)t[j+8]; }
            __bf16* dst = &sB[row][seg * 16];
            *(v8bf*)(dst)     = lo;
            *(v8bf*)(dst + 8) = hi;
        }
        if (k0 + 32 < K)  // next weight tile toward GL2 (global_prefetch_b8)
            __builtin_prefetch((const void*)(Bw + (size_t)(k0 + 32) * N + n0 + ((tid & 63) << 4)), 0, 0);
        if (!A_F32 && wv == 0)
            __builtin_amdgcn_s_wait_tensorcnt(0);
        __syncthreads();

        // ---- 8 WMMAs per wave ----
        v16bf aT[4];
        #pragma unroll
        for (int mi = 0; mi < 4; ++mi)
            aT[mi] = load_a_16x32(&sA[wrow * 64 + mi * 16][0], 40);
        v16bf bT[2];
        #pragma unroll
        for (int ni = 0; ni < 2; ++ni)
            bT[ni] = load_b_32x16(&sB[0][wcol * 32 + ni * 16], 136);
        #pragma unroll
        for (int mi = 0; mi < 4; ++mi)
            #pragma unroll
            for (int ni = 0; ni < 2; ++ni)
                acc[mi][ni] = wmma_bf16(aT[mi], bT[ni], acc[mi][ni]);
    }

    // ---- epilogue ----
    const int l  = tid & 31;
    const int cn = l & 15, hh = l >> 4;
    float*  Co = (float*) Cp;
    __bf16* Cb = (__bf16*)Cp;
    #pragma unroll
    for (int mi = 0; mi < 4; ++mi) {
        #pragma unroll
        for (int ni = 0; ni < 2; ++ni) {
            const int gcol = n0 + wcol * 32 + ni * 16 + cn;
            const float bv = bias[gcol];
            #pragma unroll
            for (int i = 0; i < 8; ++i) {
                const int grow = m0 + wrow * 64 + mi * 16 + i + 8 * hh;
                const float v = acc[mi][ni][i] + bv;
                if (OUT_F32) Co[(size_t)grow * N + gcol] = v;
                else         Cb[(size_t)grow * N + gcol] = (__bf16)v;
            }
        }
    }
}

// ---------------- RoPE: f32 (rows x 128) -> bf16, pos = row % S -------------
__global__ __launch_bounds__(64)
void rope_kernel(const float* __restrict__ in, __bf16* __restrict__ out) {
    const int r = blockIdx.x;
    const int i = threadIdx.x;
    const int s = r & (Sdim - 1);
    const float invf = __powf(10000.0f, -((2.0f * (float)i) / (float)Ddim));
    const float ang  = (float)s * invf;
    float sn, cs;  __sincosf(ang, &sn, &cs);
    const float x1 = in[(size_t)r * Ddim + i];
    const float x2 = in[(size_t)r * Ddim + 64 + i];
    out[(size_t)r * Ddim + i]      = (__bf16)(x1 * cs - x2 * sn);
    out[(size_t)r * Ddim + 64 + i] = (__bf16)(x1 * sn + x2 * cs);
}

// ---------------- Flash attention, d = 128(c) + 128(r), wave32 WMMA ---------
// Transposed scores: S^T = K @ Q^T. K's natural [key][feat] layout is the WMMA
// A-layout -> K read straight from global; Q^T staged to LDS once; V tile DMA'd
// by the TDM and overlapped with score+softmax, fenced with s_wait_tensorcnt.
// grid = (S/64, NH, B); 128 threads = 4 waves; wave owns 16 q rows.
__global__ __launch_bounds__(128)
void mla_attention_kernel(const __bf16* __restrict__ Qc,   // [B,S,H]
                          const __bf16* __restrict__ Kc,   // [B,S,H]
                          const __bf16* __restrict__ Vc,   // [B,S,H]
                          const __bf16* __restrict__ Qr,   // [B,S,D]
                          const __bf16* __restrict__ Kr,   // [B,S,D]
                          const int*    __restrict__ msk,  // [B,S]
                          __bf16* __restrict__ Ctx) {      // [B,S,H]
    __shared__ __bf16 sQt[256][72];       // [feat (c:0..127, r:128..255)][qrow]
    __shared__ __bf16 sV[64][136];        // [key][feat], TDM pad 64dw/4dw
    __shared__ __bf16 sP[4][16][72];      // per-wave P tile [qrow][key]

    const int tid = threadIdx.x;
    const int wv  = tid >> 5;
    const int l   = tid & 31;
    const int cn  = l & 15, hh = l >> 4;

    const int qt = blockIdx.x, hd = blockIdx.y, bb = blockIdx.z;
    const int qbase = qt * 64;
    const float scale = 0.08838834764831845f;   // 1/sqrt(128)

    // ---- stage Q^T once per block ----
    {
        const int qrow = tid >> 1, half = tid & 1;
        const __bf16* src = half
            ? (Qr + (size_t)(bb * Sdim + qbase + qrow) * Ddim)
            : (Qc + ((size_t)(bb * Sdim + qbase + qrow) * Hdim + hd * Ddim));
        #pragma unroll
        for (int c = 0; c < 16; ++c) {
            v8bf v = *(const v8bf*)(src + c * 8);
            #pragma unroll
            for (int e = 0; e < 8; ++e)
                sQt[half * 128 + c * 8 + e][qrow] = v[e];
        }
    }
    __syncthreads();

    // wave-private B operands (16 q columns), live for the whole kernel
    v16bf qb[8];
    #pragma unroll
    for (int ks = 0; ks < 8; ++ks)
        qb[ks] = load_b_32x16(&sQt[ks * 32][wv * 16], 72);

    const v8f vzero = {0,0,0,0,0,0,0,0};
    v8f ctx[8];
    #pragma unroll
    for (int ft = 0; ft < 8; ++ft) ctx[ft] = vzero;
    float mq = -1e30f, lq = 0.0f;       // per-lane stats for qrow = l&15

    for (int t = 0; t < Sdim / 64; ++t) {
        const int kt0 = t * 64;
        __syncthreads();                 // previous PV reads of sV complete
        if (wv == 0)                     // async V tile DMA
            tdm_load_2d_bf16((unsigned)(size_t)&sV[0][0],
                             Vc + ((size_t)(bb * Sdim + kt0) * Hdim + hd * Ddim),
                             /*w=*/128, /*h=*/64, /*stride=*/Hdim,
                             /*pad_int=*/5, /*pad_amt=*/3);

        // mask bias for this tile: 2 loads/lane, shuffled out below
        float mbv[2];
        mbv[0] = (msk[bb * Sdim + kt0 + l]      == 0) ? -1e30f : 0.0f;
        mbv[1] = (msk[bb * Sdim + kt0 + 32 + l] == 0) ? -1e30f : 0.0f;

        // ---- S^T tiles: row = key (j*16 + i + 8*hh), col = qrow (l&15) ----
        float s4[4][8];
        #pragma unroll
        for (int j = 0; j < 4; ++j) {
            const int keyrow = bb * Sdim + kt0 + j * 16 + cn;
            const __bf16* pc = Kc + ((size_t)keyrow * Hdim + hd * Ddim);
            const __bf16* pr = Kr + (size_t)keyrow * Ddim;
            v16bf ka[8];
            #pragma unroll
            for (int ks = 0; ks < 4; ++ks) ka[ks]     = load_a_global(pc, ks * 32);
            #pragma unroll
            for (int ks = 0; ks < 4; ++ks) ka[4 + ks] = load_a_global(pr, ks * 32);
            v8f st = vzero;
            #pragma unroll
            for (int ks = 0; ks < 8; ++ks) st = wmma_bf16(ka[ks], qb[ks], st);
            #pragma unroll
            for (int i = 0; i < 8; ++i) {
                const float mb = __shfl(mbv[j >> 1], (j & 1) * 16 + i + 8 * hh, 32);
                s4[j][i] = st[i] * scale + mb;
            }
        }

        // ---- online softmax (key dim = local 32 vals + half-wave combine) ----
        float rmax = -1e30f;
        #pragma unroll
        for (int j = 0; j < 4; ++j)
            #pragma unroll
            for (int i = 0; i < 8; ++i) rmax = fmaxf(rmax, s4[j][i]);
        rmax = fmaxf(rmax, __shfl_xor(rmax, 16, 32));
        const float mnew = fmaxf(mq, rmax);
        const float corr = __expf(mq - mnew);
        float p4[4][8], psum = 0.0f;
        #pragma unroll
        for (int j = 0; j < 4; ++j)
            #pragma unroll
            for (int i = 0; i < 8; ++i) {
                const float pv = (s4[j][i] <= -1e29f) ? 0.0f : __expf(s4[j][i] - mnew);
                p4[j][i] = pv; psum += pv;
            }
        psum += __shfl_xor(psum, 16, 32);
        lq = lq * corr + psum;
        mq = mnew;
        // broadcast corr from stat domain (qrow = lane) into ctx C-layout rows
        float corrb[8];
        #pragma unroll
        for (int i = 0; i < 8; ++i) corrb[i] = __shfl(corr, i + 8 * hh, 32);
        #pragma unroll
        for (int ft = 0; ft < 8; ++ft)
            #pragma unroll
            for (int i = 0; i < 8; ++i) ctx[ft][i] *= corrb[i];

        // ---- P^T -> LDS as [qrow][key] (A-layout source for PV) ----
        #pragma unroll
        for (int j = 0; j < 4; ++j)
            #pragma unroll
            for (int i = 0; i < 8; ++i)
                sP[wv][cn][j * 16 + i + 8 * hh] = (__bf16)p4[j][i];

        if (wv == 0) __builtin_amdgcn_s_wait_tensorcnt(0);
        __syncthreads();                 // sV ready for everyone

        // ---- ctx += P @ V, 16 WMMAs ----
        v16bf pa[2];
        #pragma unroll
        for (int ks = 0; ks < 2; ++ks)
            pa[ks] = load_a_16x32(&sP[wv][0][ks * 32], 72);
        #pragma unroll
        for (int ft = 0; ft < 8; ++ft)
            #pragma unroll
            for (int ks = 0; ks < 2; ++ks) {
                v16bf vb = load_b_32x16(&sV[ks * 32][ft * 16], 136);
                ctx[ft] = wmma_bf16(pa[ks], vb, ctx[ft]);
            }
    }

    // ---- normalize + write ----
    float linv[8];
    #pragma unroll
    for (int i = 0; i < 8; ++i) linv[i] = 1.0f / __shfl(lq, i + 8 * hh, 32);
    #pragma unroll
    for (int ft = 0; ft < 8; ++ft) {
        #pragma unroll
        for (int i = 0; i < 8; ++i) {
            const int qrow = qbase + wv * 16 + i + 8 * hh;
            const int col  = hd * Ddim + ft * 16 + cn;
            Ctx[(size_t)(bb * Sdim + qrow) * Hdim + col] = (__bf16)(ctx[ft][i] * linv[i]);
        }
    }
}

// ---------------- host orchestration ----------------
extern "C" void kernel_launch(void* const* d_in, const int* in_sizes, int n_in,
                              void* d_out, int out_size, void* d_ws, size_t ws_size,
                              hipStream_t stream) {
    (void)in_sizes; (void)n_in; (void)out_size; (void)ws_size;
    const float* X     = (const float*)d_in[0];
    const int*   amask = (const int*)  d_in[1];
    const float* W_DKV = (const float*)d_in[2];  const float* b_DKV = (const float*)d_in[3];
    const float* W_DQ  = (const float*)d_in[4];  const float* b_DQ  = (const float*)d_in[5];
    const float* W_UK  = (const float*)d_in[6];  const float* b_UK  = (const float*)d_in[7];
    const float* W_UV  = (const float*)d_in[8];  const float* b_UV  = (const float*)d_in[9];
    const float* W_UQ  = (const float*)d_in[10]; const float* b_UQ  = (const float*)d_in[11];
    const float* W_KR  = (const float*)d_in[12]; const float* b_KR  = (const float*)d_in[13];
    const float* W_QR  = (const float*)d_in[14]; const float* b_QR  = (const float*)d_in[15];
    const float* W_O   = (const float*)d_in[16]; const float* b_O   = (const float*)d_in[17];

    char* ws = (char*)d_ws;
    size_t off = 0;
    auto alloc = [&](size_t bytes) { size_t o = off; off = (off + bytes + 255) & ~(size_t)255; return o; };
    __bf16* ckv   = (__bf16*)(ws + alloc((size_t)Mrows * LATD * 2));
    __bf16* cq    = (__bf16*)(ws + alloc((size_t)Mrows * LATD * 2));
    __bf16* kc    = (__bf16*)(ws + alloc((size_t)Mrows * Hdim * 2));
    __bf16* vc    = (__bf16*)(ws + alloc((size_t)Mrows * Hdim * 2));
    __bf16* qc    = (__bf16*)(ws + alloc((size_t)Mrows * Hdim * 2));
    float*  krraw = (float*) (ws + alloc((size_t)Mrows * Ddim * 4));
    float*  qrraw = (float*) (ws + alloc((size_t)Mrows * Ddim * 4));
    __bf16* krb   = (__bf16*)(ws + alloc((size_t)Mrows * Ddim * 2));
    __bf16* qrb   = (__bf16*)(ws + alloc((size_t)Mrows * Ddim * 2));
    __bf16* ctx   = (__bf16*)(ws + alloc((size_t)Mrows * Hdim * 2));

    const dim3 blk(256);
    gemm_wmma_kernel<true, false><<<dim3(LATD/128, Mrows/128), blk, 0, stream>>>(
        X, W_DKV, b_DKV, ckv, Mrows, LATD, Hdim);
    gemm_wmma_kernel<true, false><<<dim3(LATD/128, Mrows/128), blk, 0, stream>>>(
        X, W_DQ, b_DQ, cq, Mrows, LATD, Hdim);
    gemm_wmma_kernel<false, false><<<dim3(Hdim/128, Mrows/128), blk, 0, stream>>>(
        ckv, W_UK, b_UK, kc, Mrows, Hdim, LATD);
    gemm_wmma_kernel<false, false><<<dim3(Hdim/128, Mrows/128), blk, 0, stream>>>(
        ckv, W_UV, b_UV, vc, Mrows, Hdim, LATD);
    gemm_wmma_kernel<false, false><<<dim3(Hdim/128, Mrows/128), blk, 0, stream>>>(
        cq, W_UQ, b_UQ, qc, Mrows, Hdim, LATD);
    gemm_wmma_kernel<true, true><<<dim3(Ddim/128, Mrows/128), blk, 0, stream>>>(
        X, W_KR, b_KR, krraw, Mrows, Ddim, Hdim);
    gemm_wmma_kernel<true, true><<<dim3(Ddim/128, Mrows/128), blk, 0, stream>>>(
        X, W_QR, b_QR, qrraw, Mrows, Ddim, Hdim);
    rope_kernel<<<dim3(Mrows), dim3(64), 0, stream>>>(krraw, krb);
    rope_kernel<<<dim3(Mrows), dim3(64), 0, stream>>>(qrraw, qrb);
    mla_attention_kernel<<<dim3(Sdim/64, NHEAD, Bdim), dim3(128), 0, stream>>>(
        qc, kc, vc, qrb, krb, amask, ctx);
    gemm_wmma_kernel<false, true><<<dim3(Hdim/128, Mrows/128), blk, 0, stream>>>(
        ctx, W_O, b_O, (float*)d_out, Mrows, Hdim, Hdim);
}
